// mula_vgg_based_network_6786048328577
// MI455X (gfx1250) — matrix operand order repaired and checked by reading, not compiled, not run on hardware
//
#include <hip/hip_runtime.h>
#include <hip/hip_bf16.h>

// ---------------------------------------------------------------------------
// Types for CDNA5 WMMA (wave32): D(f32 16x16) = A(bf16 16x32) x B(bf16 32x16) + C
// ---------------------------------------------------------------------------
typedef __attribute__((ext_vector_type(16))) __bf16 v16bf;
typedef __attribute__((ext_vector_type(8)))  float  v8f;

union FragAB {
    unsigned int u[8];
    v16bf        bf;
};

__device__ __forceinline__ unsigned short f2bf(float f) {
    unsigned int u = __float_as_uint(f);
    u = u + 0x7FFFu + ((u >> 16) & 1u);      // round-to-nearest-even
    return (unsigned short)(u >> 16);
}
__device__ __forceinline__ float bf2f(unsigned short h) {
    return __uint_as_float(((unsigned int)h) << 16);
}

// Async global->LDS copy (CDNA5 GLOBAL_LOAD_ASYNC_TO_LDS_B128, ASYNCcnt-tracked).
// Per CDNA5 ISA: the immediate offset is added to both the LDS and the global address.
__device__ __forceinline__ void async_lds_b128(unsigned lds_addr, const void* gaddr) {
    asm volatile("global_load_async_to_lds_b128 %0, %1, off"
                 :: "v"(lds_addr), "v"(gaddr) : "memory");
}
__device__ __forceinline__ void async_lds_b128_off16(unsigned lds_addr, const void* gaddr) {
    asm volatile("global_load_async_to_lds_b128 %0, %1, off offset:16"
                 :: "v"(lds_addr), "v"(gaddr) : "memory");
}
__device__ __forceinline__ void wait_asynccnt0() {
    asm volatile("s_wait_asynccnt 0x0" ::: "memory");
}

// ---------------------------------------------------------------------------
// f32 -> bf16 conversion (plain)
// ---------------------------------------------------------------------------
__global__ void f32_to_bf16_k(const float* __restrict__ in,
                              unsigned short* __restrict__ out, long n) {
    long i = (long)blockIdx.x * blockDim.x + threadIdx.x;
    if (i < n) out[i] = f2bf(in[i]);
}

// ---------------------------------------------------------------------------
// Weight conversion with K-padding: [OC,Ktot] f32 -> [OC,Kpad] bf16, zero tail.
// Kpad is a multiple of 64 so the GEMM kernel needs no K guards and can use
// aligned b128 / async-to-LDS transfers.
// ---------------------------------------------------------------------------
__global__ void cvt_w_pad(const float* __restrict__ in,
                          unsigned short* __restrict__ out,
                          int OC, int Ktot, int Kpad) {
    long i = (long)blockIdx.x * blockDim.x + threadIdx.x;
    long total = (long)OC * Kpad;
    if (i >= total) return;
    int k  = (int)(i % Kpad);
    int oc = (int)(i / Kpad);
    out[i] = (k < Ktot) ? f2bf(in[(long)oc * Ktot + k]) : (unsigned short)0;
}

// ---------------------------------------------------------------------------
// Implicit-GEMM convolution with WMMA (bf16 in/out, f32 accumulate).
// Block = 256 threads = 8 wave32s.  Block tile: 64 OC x 64 pixels, K staged 64.
// Wave (mt = w>>1, np = w&1) computes M rows [mt*16,+16) x N cols [np*32,+32)
// as two 16x16 tiles -> 2 accumulators, 4 v_wmma per K stage.
// A (weights) is staged with async global->LDS b128 copies (ASYNCcnt), B
// (im2col) with packed ds_write_b32; s_wait_asynccnt 0 precedes the barrier.
// ---------------------------------------------------------------------------
#define BM 64
#define BN 64
#define BK 64

__global__ __launch_bounds__(256)
void conv_wmma_bf16(const unsigned short* __restrict__ in,   // bf16 [B,IC,H,W]
                    const unsigned short* __restrict__ wt,   // bf16 [OC, Kpad]
                    const float* __restrict__ bias,          // f32  [OC]
                    unsigned short* __restrict__ out,        // bf16 [B,OC,OH,OW]
                    int Bn, int IC, int OC, int H, int W,
                    int OH, int OW, int KH, int KW,
                    int stride, int pad, int relu, int Kpad) {
    __shared__ unsigned short Alds[BM][BK + 8];  // OC rows, K contiguous
    __shared__ unsigned short Blds[BN][BK + 8];  // pixel rows, K contiguous

    const int tid  = threadIdx.x;
    const int lane = tid & 31;
    const int wave = tid >> 5;      // 0..7
    const int mt   = wave >> 1;     // 0..3  M sub-tile
    const int np   = wave & 1;      // 0..1  N half (2 tiles each)

    const int ocBase  = blockIdx.y * BM;
    const int pixBase = blockIdx.x * BN;
    const int KK   = KH * KW;
    const int Ktot = IC * KK;
    const int HW   = H * W;
    const int OHW  = OH * OW;
    const int Npix = Bn * OHW;

    // ---- per-thread staging coordinates: fixed for the whole kernel ----
    const int srow  = tid >> 2;        // 0..63 : A row (oc) and B row (pixel)
    const int kbase = (tid & 3) * 16;  // 0,16,32,48

    // decode this thread's output pixel ONCE (no divisions in the K loop)
    const int  gp     = pixBase + srow;
    const bool pvalid = (gp < Npix);
    int b = 0, oh = 0, ow = 0;
    if (pvalid) {
        b = gp / OHW; int sp = gp - b * OHW;
        oh = sp / OW; ow = sp - oh * OW;
    }
    const int  ohs    = oh * stride - pad;
    const int  ows    = ow * stride - pad;
    const long inBase = (long)b * IC * HW;
    const long arow   = (long)(ocBase + srow) * Kpad;

    // LDS byte address of this thread's A-staging slot (for async-to-LDS)
    const unsigned aldsAddr =
        (unsigned)(unsigned long long)&Alds[srow][kbase];

    v8f acc[2];
    acc[0] = (v8f){0.f, 0.f, 0.f, 0.f, 0.f, 0.f, 0.f, 0.f};
    acc[1] = (v8f){0.f, 0.f, 0.f, 0.f, 0.f, 0.f, 0.f, 0.f};

    for (int k0 = 0; k0 < Kpad; k0 += BK) {
        // Prefetch next weight chunk (global_prefetch_b8)
        if (k0 + BK < Kpad)
            __builtin_prefetch(wt + arow + k0 + BK, 0, 1);

        // ---- stage A: 32 bytes of weights via two async global->LDS b128 ----
        {
            const unsigned short* gsrc = wt + arow + k0 + kbase;
            async_lds_b128(aldsAddr, gsrc);           // bytes [0,16)
            async_lds_b128_off16(aldsAddr, gsrc);     // bytes [16,32)
        }
        // ---- stage B: 16 consecutive-K im2col taps, incremental (ic,r,s) ----
        {
            int gk = k0 + kbase;
            int ic, r_, s_;
            if (KK > 1) {            // one div pair per stage, then pure adds
                ic = gk / KK; int rs = gk - ic * KK;
                r_ = rs / KW; s_ = rs - r_ * KW;
            } else { ic = gk; r_ = 0; s_ = 0; }
            unsigned int* dst = (unsigned int*)&Blds[srow][kbase];
            #pragma unroll
            for (int j = 0; j < 8; ++j) {
                unsigned int pk = 0;
                #pragma unroll
                for (int h2 = 0; h2 < 2; ++h2) {
                    unsigned short v = 0;
                    if (pvalid && gk < Ktot) {
                        int ih = ohs + r_, iw = ows + s_;
                        if (ih >= 0 && ih < H && iw >= 0 && iw < W)
                            v = in[inBase + (long)ic * HW + ih * W + iw];
                    }
                    pk |= ((unsigned int)v) << (16 * h2);
                    ++gk;
                    if (++s_ == KW) { s_ = 0; if (++r_ == KH) { r_ = 0; ++ic; } }
                }
                dst[j] = pk;
            }
        }
        wait_asynccnt0();        // A-copy landed in LDS before the barrier
        __syncthreads();

        // ---- compute: 2 K-subchunks x (1 A frag, 2 B frags) = 4 v_wmma ----
        const int hi   = (lane >= 16);
        const int row  = mt * 16 + (lane & 15);
        const int col0 = np * 32 + (lane & 15);
        #pragma unroll
        for (int ks = 0; ks < 2; ++ks) {
            const int kb = ks * 32;
            FragAB fa, fb0, fb1;
            #pragma unroll
            for (int v = 0; v < 8; ++v) {
                int kk = kb + (hi ? 16 : 0) + 2 * v;
                fa.u[v] = *(const unsigned int*)&Alds[row][kk];
            }
            #pragma unroll
            for (int v = 0; v < 8; ++v) {
                int kk = kb + (hi ? 16 : 0) + 2 * v;
                fb0.u[v] = *(const unsigned int*)&Blds[col0][kk];
                fb1.u[v] = *(const unsigned int*)&Blds[col0 + 16][kk];
            }
            acc[0] = __builtin_amdgcn_wmma_f32_16x16x32_bf16(
                         false, fa.bf, false, fb0.bf, (short)0, acc[0], false, false);
            acc[1] = __builtin_amdgcn_wmma_f32_16x16x32_bf16(
                         false, fa.bf, false, fb1.bf, (short)0, acc[1], false, false);
        }
        __syncthreads();
    }

    // ---- epilogue: bias + ReLU + bf16 store ----
    const int hi  = (lane >= 16);
    const int ocT = ocBase + mt * 16;
    #pragma unroll
    for (int half = 0; half < 2; ++half) {
        int n = pixBase + np * 32 + half * 16 + (lane & 15);
        if (n < Npix) {
            int nb = n / OHW;
            int sp = n - nb * OHW;
            #pragma unroll
            for (int v = 0; v < 8; ++v) {
                int m  = hi ? (v + 8) : v;
                int oc = ocT + m;
                if (oc < OC) {
                    float r = acc[half][v] + bias[oc];
                    if (relu && r < 0.f) r = 0.f;
                    out[((long)(nb * OC + oc)) * OHW + sp] = f2bf(r);
                }
            }
        }
    }
}

// ---------------------------------------------------------------------------
// 2x2 max-pool, stride 2, VALID (bf16)
// ---------------------------------------------------------------------------
__global__ void maxpool2_bf16(const unsigned short* __restrict__ in,
                              unsigned short* __restrict__ out,
                              int BC, int H, int W) {
    int OHc = H >> 1, OWc = W >> 1;
    long n = (long)blockIdx.x * blockDim.x + threadIdx.x;
    long total = (long)BC * OHc * OWc;
    if (n >= total) return;
    int ow = (int)(n % OWc); long t = n / OWc;
    int oh = (int)(t % OHc); int bc = (int)(t / OHc);
    const unsigned short* p = in + ((long)bc * H + oh * 2) * W + ow * 2;
    float m = bf2f(p[0]);
    m = fmaxf(m, bf2f(p[1]));
    m = fmaxf(m, bf2f(p[W]));
    m = fmaxf(m, bf2f(p[W + 1]));
    out[n] = f2bf(m);
}

// ---------------------------------------------------------------------------
// Dynamic per-(B,C) depthwise 3x3 conv, pad 1 (bf16)
// ---------------------------------------------------------------------------
__global__ void adaptive_conv_bf16(const unsigned short* __restrict__ x,
                                   const unsigned short* __restrict__ theta, // [BC,3,3]
                                   unsigned short* __restrict__ out,
                                   int BC, int H, int W) {
    long n = (long)blockIdx.x * blockDim.x + threadIdx.x;
    long total = (long)BC * H * W;
    if (n >= total) return;
    int w = (int)(n % W); long t = n / W;
    int h = (int)(t % H); int bc = (int)(t / H);
    const unsigned short* tp = theta + (long)bc * 9;
    const unsigned short* xp = x + (long)bc * H * W;
    float acc = 0.f;
    #pragma unroll
    for (int r = 0; r < 3; ++r) {
        int ih = h - 1 + r;
        if (ih < 0 || ih >= H) continue;
        #pragma unroll
        for (int s = 0; s < 3; ++s) {
            int iw = w - 1 + s;
            if (iw < 0 || iw >= W) continue;
            acc += bf2f(xp[ih * W + iw]) * bf2f(tp[r * 3 + s]);
        }
    }
    out[n] = f2bf(acc);
}

// ---------------------------------------------------------------------------
// Elementwise add (bf16)
// ---------------------------------------------------------------------------
__global__ void add_bf16(const unsigned short* __restrict__ a,
                         const unsigned short* __restrict__ b,
                         unsigned short* __restrict__ c, long n) {
    long i = (long)blockIdx.x * blockDim.x + threadIdx.x;
    if (i < n) c[i] = f2bf(bf2f(a[i]) + bf2f(b[i]));
}

// ---------------------------------------------------------------------------
// 1x1 classifier conv: bf16 features, f32 weights/bias, f32 out
// ---------------------------------------------------------------------------
__global__ void conv1x1_f32(const unsigned short* __restrict__ feat, // [B,IC,HW]
                            const float* __restrict__ w,             // [OC,IC]
                            const float* __restrict__ bias,
                            float* __restrict__ out,                 // [B,OC,HW]
                            int Bn, int IC, int OC, int HWc) {
    long t = (long)blockIdx.x * blockDim.x + threadIdx.x;
    long total = (long)Bn * OC * HWc;
    if (t >= total) return;
    int sp = (int)(t % HWc); long r = t / HWc;
    int oc = (int)(r % OC);  int b  = (int)(r / OC);
    const unsigned short* f = feat + (long)b * IC * HWc + sp;
    const float* wr = w + (long)oc * IC;
    float s = bias[oc];
    for (int ic = 0; ic < IC; ++ic)
        s += bf2f(f[(long)ic * HWc]) * wr[ic];
    out[t] = s;
}

// ---------------------------------------------------------------------------
// 8x bilinear upsample (half-pixel centers), f32
// ---------------------------------------------------------------------------
__global__ void upsample8_f32(const float* __restrict__ in, float* __restrict__ out,
                              int BC, int IH, int IW) {
    int OHs = IH * 8, OWs = IW * 8;
    long n = (long)blockIdx.x * blockDim.x + threadIdx.x;
    long total = (long)BC * OHs * OWs;
    if (n >= total) return;
    int ox = (int)(n % OWs); long t = n / OWs;
    int oy = (int)(t % OHs); int bc = (int)(t / OHs);
    float sy = (oy + 0.5f) * 0.125f - 0.5f;
    float sx = (ox + 0.5f) * 0.125f - 0.5f;
    int y0 = (int)floorf(sy), x0 = (int)floorf(sx);
    float fy = sy - y0, fx = sx - x0;
    int y1 = y0 + 1, x1 = x0 + 1;
    y0 = min(max(y0, 0), IH - 1); y1 = min(max(y1, 0), IH - 1);
    x0 = min(max(x0, 0), IW - 1); x1 = min(max(x1, 0), IW - 1);
    const float* p = in + (long)bc * IH * IW;
    float v00 = p[y0 * IW + x0], v01 = p[y0 * IW + x1];
    float v10 = p[y1 * IW + x0], v11 = p[y1 * IW + x1];
    float v0 = v00 + (v01 - v00) * fx;
    float v1 = v10 + (v11 - v10) * fx;
    out[n] = v0 + (v1 - v0) * fy;
}

// ---------------------------------------------------------------------------
// In-place log_softmax over channel axis, f32 [B,C,HW]
// ---------------------------------------------------------------------------
__global__ void log_softmax_ch(float* __restrict__ data, int Bn, int C, int HWs) {
    long n = (long)blockIdx.x * blockDim.x + threadIdx.x;
    long total = (long)Bn * HWs;
    if (n >= total) return;
    int sp = (int)(n % HWs); int b = (int)(n / HWs);
    float* p = data + (long)b * C * HWs + sp;
    float mx = -3.4e38f;
    for (int c = 0; c < C; ++c) mx = fmaxf(mx, p[(long)c * HWs]);
    float s = 0.f;
    for (int c = 0; c < C; ++c) s += expf(p[(long)c * HWs] - mx);
    float lse = mx + logf(s);
    for (int c = 0; c < C; ++c) p[(long)c * HWs] -= lse;
}

// ===========================================================================
// Host orchestration
// ===========================================================================
extern "C" void kernel_launch(void* const* d_in, const int* in_sizes, int n_in,
                              void* d_out, int out_size, void* d_ws, size_t ws_size,
                              hipStream_t stream) {
    (void)in_sizes; (void)n_in; (void)out_size; (void)ws_size;

    const int Bn = 4;
    auto cdiv = [](long a, long b) { return (a + b - 1) / b; };

    // ---- workspace bump allocator ----
    size_t off = 0;
    auto alloc = [&](size_t bytes) -> void* {
        off = (off + 255) & ~(size_t)255;
        void* p = (char*)d_ws + off;
        off += bytes;
        return p;
    };
    unsigned short* xbf       = (unsigned short*)alloc(602112ull * 2);     // 4x3x224x224
    unsigned short* bufA      = (unsigned short*)alloc(12845056ull * 2);   // 4x64x224x224
    unsigned short* bufB      = (unsigned short*)alloc(12845056ull * 2);
    unsigned short* featPose  = (unsigned short*)alloc(1605632ull * 2);    // 4x512x28x28
    unsigned short* featPars  = (unsigned short*)alloc(1605632ull * 2);
    unsigned short* thetaPose = (unsigned short*)alloc(18432ull * 2);      // 4x512x3x3
    unsigned short* thetaPars = (unsigned short*)alloc(18432ull * 2);
    unsigned short* wslot     = (unsigned short*)alloc(2359296ull * 2);    // max OC*Kpad
    float*          ppF32     = (float*)alloc(62720ull * 4);               // 4x20x28x28

    auto cvt = [&](const float* src, unsigned short* dst, long n) {
        f32_to_bf16_k<<<(unsigned)cdiv(n, 256), 256, 0, stream>>>(src, dst, n);
    };
    // convert + K-pad the layer weights into wslot, then launch the WMMA conv
    auto conv = [&](const unsigned short* in, const float* Wf, const float* Bf,
                    unsigned short* out, int ic, int oc, int h, int wd,
                    int oh, int ow, int kh, int stride, int pad, int relu) {
        int KK = kh * kh, Ktot = ic * KK;
        int Kpad = (int)(((long)Ktot + 63) / 64) * 64;
        long wn = (long)oc * Kpad;
        cvt_w_pad<<<(unsigned)cdiv(wn, 256), 256, 0, stream>>>(Wf, wslot, oc, Ktot, Kpad);
        long npix = (long)Bn * oh * ow;
        dim3 grid((unsigned)cdiv(npix, BN), (unsigned)cdiv(oc, BM));
        conv_wmma_bf16<<<grid, 256, 0, stream>>>(in, wslot, Bf, out, Bn, ic, oc,
                                                 h, wd, oh, ow, kh, kh, stride, pad,
                                                 relu, Kpad);
    };
    auto pool = [&](const unsigned short* in, unsigned short* out, int bc, int h, int wd) {
        long total = (long)bc * (h >> 1) * (wd >> 1);
        maxpool2_bf16<<<(unsigned)cdiv(total, 256), 256, 0, stream>>>(in, out, bc, h, wd);
    };

    const float* X = (const float*)d_in[0];
    cvt(X, xbf, 602112);

    // ---- VGG configuration ----
    static const int vIC[13] = {3, 64, 64, 128, 128, 256, 256, 256, 512, 512, 512, 512, 512};
    static const int vOC[13] = {64, 64, 128, 128, 256, 256, 256, 512, 512, 512, 512, 512, 512};
    static const int vH [13] = {224, 224, 112, 112, 56, 56, 56, 28, 28, 28, 28, 28, 28};

    auto run_vgg = [&](const unsigned short* xin, int base, unsigned short* featOut) {
        const unsigned short* cur = xin;
        unsigned short* bufs[2] = {bufA, bufB};
        int pp = 0;
        for (int l = 0; l < 13; ++l) {
            const float* Wf = (const float*)d_in[base + 2 * l];
            const float* Bf = (const float*)d_in[base + 2 * l + 1];
            int ic = vIC[l], oc = vOC[l], h = vH[l];
            bool pl = (l == 1 || l == 3 || l == 6);
            unsigned short* dst = (!pl && l == 12) ? featOut : bufs[pp];
            conv(cur, Wf, Bf, dst, ic, oc, h, h, h, h, 3, 1, 1, /*relu*/1);
            cur = dst; pp ^= 1;
            if (pl) {
                unsigned short* pd = bufs[pp];
                pool(cur, pd, Bn * oc, h, h);
                cur = pd; pp ^= 1;
            }
        }
    };

    run_vgg(xbf, 1, featPose);    // pose_enc   : d_in[1..26]
    run_vgg(xbf, 27, featPars);   // parsing_enc: d_in[27..52]

    // ---- adapters: conv s2 p0 (28->13), pool (->6), conv p1, pool (->3), conv p1 ----
    auto run_adapter = [&](const unsigned short* feat, int base, unsigned short* thOut) {
        conv(feat, (const float*)d_in[base + 0], (const float*)d_in[base + 1],
             bufA, 512, 512, 28, 28, 13, 13, 3, 2, 0, 0);
        pool(bufA, bufB, Bn * 512, 13, 13);                                  // -> 6x6
        conv(bufB, (const float*)d_in[base + 2], (const float*)d_in[base + 3],
             bufA, 512, 512, 6, 6, 6, 6, 3, 1, 1, 0);
        pool(bufA, bufB, Bn * 512, 6, 6);                                    // -> 3x3
        conv(bufB, (const float*)d_in[base + 4], (const float*)d_in[base + 5],
             thOut, 512, 512, 3, 3, 3, 3, 3, 1, 1, 0);
    };
    run_adapter(featPars, 53, thetaPose);   // pose_theta  = adapter(parsing_feat)
    run_adapter(featPose, 59, thetaPars);   // parsing_phi = adapter(pose_feat)

    const long NFEAT = 1605632;             // 4x512x28x28
    float* outF = (float*)d_out;

    // ---- pose refinement: U -> adaptive -> V(relu) -> +feat -> cls ----
    conv(featPose, (const float*)d_in[65], (const float*)d_in[66],
         bufA, 512, 512, 28, 28, 28, 28, 1, 1, 0, 0);
    adaptive_conv_bf16<<<(unsigned)cdiv(NFEAT, 256), 256, 0, stream>>>(bufA, thetaPose, bufB,
                                                                       Bn * 512, 28, 28);
    conv(bufB, (const float*)d_in[67], (const float*)d_in[68],
         bufA, 512, 512, 28, 28, 28, 28, 1, 1, 0, 1);
    add_bf16<<<(unsigned)cdiv(NFEAT, 256), 256, 0, stream>>>(featPose, bufA, bufB, NFEAT);
    {   // pose_cls: 512 -> 17, f32 out directly into d_out[0 : 53312]
        long total = (long)Bn * 17 * 784;
        conv1x1_f32<<<(unsigned)cdiv(total, 256), 256, 0, stream>>>(
            bufB, (const float*)d_in[73], (const float*)d_in[74], outF, Bn, 512, 17, 784);
    }

    // ---- parsing refinement ----
    conv(featPars, (const float*)d_in[69], (const float*)d_in[70],
         bufA, 512, 512, 28, 28, 28, 28, 1, 1, 0, 0);
    adaptive_conv_bf16<<<(unsigned)cdiv(NFEAT, 256), 256, 0, stream>>>(bufA, thetaPars, bufB,
                                                                       Bn * 512, 28, 28);
    conv(bufB, (const float*)d_in[71], (const float*)d_in[72],
         bufA, 512, 512, 28, 28, 28, 28, 1, 1, 0, 1);
    add_bf16<<<(unsigned)cdiv(NFEAT, 256), 256, 0, stream>>>(featPars, bufA, bufB, NFEAT);
    {   // parsing_cls: 512 -> 20, f32
        long total = (long)Bn * 20 * 784;
        conv1x1_f32<<<(unsigned)cdiv(total, 256), 256, 0, stream>>>(
            bufB, (const float*)d_in[75], (const float*)d_in[76], ppF32, Bn, 512, 20, 784);
    }

    // ---- 8x bilinear upsample + log_softmax into d_out[53312 : 53312+4014080] ----
    float* parsOut = outF + 53312;
    {
        long total = (long)Bn * 20 * 224 * 224;
        upsample8_f32<<<(unsigned)cdiv(total, 256), 256, 0, stream>>>(ppF32, parsOut,
                                                                     Bn * 20, 28, 28);
        long px = (long)Bn * 224 * 224;
        log_softmax_ch<<<(unsigned)cdiv(px, 256), 256, 0, stream>>>(parsOut, Bn, 20,
                                                                    224 * 224);
    }
}
